// DecoderBlockMAskedSelfAttention_82738249990895
// MI455X (gfx1250) — matrix operand order repaired and checked by reading, not compiled, not run on hardware
//
#include <hip/hip_runtime.h>
#include <math.h>

// Problem constants (reference: B=2, T=2048, C=1024, H=16, HD=64, FF=2048)
#define Bsz   2
#define Tn    2048
#define Cn    1024
#define Hn    16
#define HDn   64
#define FFn   2048
#define Mrows (Bsz * Tn)   // 4096 tokens

typedef __attribute__((ext_vector_type(16))) __bf16        bvec16;
typedef __attribute__((ext_vector_type(8)))  float         fvec8;
typedef __attribute__((ext_vector_type(4)))  unsigned int  u32x4;
typedef __attribute__((ext_vector_type(4)))  unsigned int  v4u;
typedef __attribute__((ext_vector_type(8)))  int           v8i_t;
typedef __attribute__((ext_vector_type(4)))  int           v4i_t;

union BFrag { bvec16 v; u32x4 u[2]; };

__device__ __forceinline__ unsigned short f2bf(float f) {
  unsigned int u = __float_as_uint(f);
  u += 0x7FFFu + ((u >> 16) & 1u);   // round-to-nearest-even
  return (unsigned short)(u >> 16);
}

// ---------------------------------------------------------------- f32 -> bf16
__global__ void cvt_f32_bf16(const float* __restrict__ in,
                             unsigned short* __restrict__ out, int n) {
  int i = blockIdx.x * blockDim.x + threadIdx.x;
  if (i < n) out[i] = f2bf(in[i]);
}

// ---------------------------------------------------------------- RMSNorm -> bf16
__global__ __launch_bounds__(256)
void rmsnorm_bf16(const float* __restrict__ x, const float* __restrict__ g,
                  unsigned short* __restrict__ out) {
  __shared__ float red[8];
  __shared__ float inv_s;
  int row = blockIdx.x;
  const float* xr = x + (size_t)row * Cn;
  float ss = 0.f;
  for (int i = threadIdx.x; i < Cn; i += 256) { float v = xr[i]; ss += v * v; }
  #pragma unroll
  for (int off = 16; off > 0; off >>= 1) ss += __shfl_down(ss, off, 32);
  int wave = threadIdx.x >> 5, lane = threadIdx.x & 31;
  if (lane == 0) red[wave] = ss;
  __syncthreads();
  if (threadIdx.x == 0) {
    float t = 0.f;
    #pragma unroll
    for (int i = 0; i < 8; i++) t += red[i];
    inv_s = rsqrtf(t / (float)Cn + 1.1920929e-07f);   // finfo(f32).eps
  }
  __syncthreads();
  float inv = inv_s;
  for (int i = threadIdx.x; i < Cn; i += 256)
    out[(size_t)row * Cn + i] = f2bf(xr[i] * inv * g[i]);
}

// ---------------------------------------------------------------- V transpose: [b,t,h,d] -> [b,h,d,t]
__global__ void transpose_v(const unsigned short* __restrict__ v,
                            unsigned short* __restrict__ vT) {
  int idx = blockIdx.x * 256 + threadIdx.x;
  if (idx >= Mrows * Cn) return;
  int tok = idx / Cn, c = idx % Cn;
  int b = tok / Tn, t = tok % Tn;
  int h = c / HDn, d = c % HDn;
  vT[((size_t)((b * Hn + h) * HDn + d)) * Tn + t] = v[idx];
}

// ---------------------------------------------------------------- TDM helper: queue W tile [256 rows x 32 cols] -> LDS
// LDS rows padded by TDM hardware: pad 4 DWORDs every 16 DWORDs -> 80B row stride.
#define WT_ROW 40   // padded LDS row stride in bf16 elements (80 bytes)

__device__ __forceinline__ void tdm_load_wtile(const unsigned short* W, int Kdim,
                                               int nblk, int k,
                                               unsigned short* lds_dst) {
  unsigned long long ga =
      (unsigned long long)(const void*)W + ((size_t)nblk * Kdim + k) * 2ull;
  unsigned int lds = (unsigned int)(unsigned long long)(void*)lds_dst;  // LDS byte offset
  v4u g0;
  g0[0] = 1u;                                     // count=1 (valid descriptor)
  g0[1] = lds;                                    // lds_addr
  g0[2] = (unsigned int)(ga & 0xFFFFFFFFull);     // global_addr[31:0]
  g0[3] = (unsigned int)((ga >> 32) & 0x01FFFFFFull) | (2u << 30);  // addr[56:32] | type=2
  v8i_t g1;
  g1[0] = (1 << 16)        // data_size = 2 bytes
        | (1 << 20)        // pad_enable
        | (3 << 22)        // pad_interval: 16 DWORDs
        | (3 << 25);       // pad_amount: 4 DWORDs
  g1[1] = (int)(32u << 16);   // tensor_dim0 = 32 (lo16)
  g1[2] = (int)(256u << 16);  // tensor_dim0 hi = 0 | tensor_dim1 = 256 (lo16)
  g1[3] = (int)(32u << 16);   // tensor_dim1 hi = 0 | tile_dim0 = 32
  g1[4] = 256;                // tile_dim1 = 256, tile_dim2 = 0
  g1[5] = Kdim;               // tensor_dim0_stride[31:0]
  g1[6] = 0;
  g1[7] = 0;
  v4i_t g2 = {0, 0, 0, 0};
  v4i_t g3 = {0, 0, 0, 0};
#if defined(__clang_major__) && (__clang_major__ >= 23)
  v8i_t gz = {0, 0, 0, 0, 0, 0, 0, 0};
  __builtin_amdgcn_tensor_load_to_lds(g0, g1, g2, g3, gz, 0);
#else
  __builtin_amdgcn_tensor_load_to_lds(g0, g1, g2, g3, 0);
#endif
}

// ---------------------------------------------------------------- WMMA GEMM: C[M,N] = A[M,K] @ W[N,K]^T (+epilogue)
// Block tile 64Mx256N, wave tile 32Mx64N. Shared W K-tile staged by the Tensor
// Data Mover into double-buffered LDS (issued by wave 0, TENSORcnt-tracked),
// overlapped with WMMA compute on the previous tile.
// EPI 0: store bf16   EPI 1: f32 store = resid + acc + bias   EPI 2: bf16 gelu(acc+bias)
template <int EPI>
__global__ __launch_bounds__(256)
void gemm_bf16_tdm(const unsigned short* __restrict__ A,
                   const unsigned short* __restrict__ W,
                   const float* __restrict__ bias,
                   const float* __restrict__ resid,
                   void* __restrict__ out, int Ndim, int Kdim) {
  __shared__ unsigned short wtile[2][256 * WT_ROW];   // 2 x 20 KB
  int wave = threadIdx.x >> 5, lane = threadIdx.x & 31;
  int hg = lane >> 4, ln = lane & 15;
  int wm = wave & 1, wn = wave >> 1;
  int mbase = blockIdx.x * 64 + wm * 32;
  int nblk  = blockIdx.y * 256;

  fvec8 acc[2][4];
  #pragma unroll
  for (int mt = 0; mt < 2; mt++)
    #pragma unroll
    for (int f = 0; f < 4; f++)
      #pragma unroll
      for (int r = 0; r < 8; r++) acc[mt][f][r] = 0.f;

  const unsigned short* arow0 = A + (size_t)(mbase + ln) * Kdim + 8 * hg;
  const unsigned short* arow1 = arow0 + (size_t)16 * Kdim;

  if (wave == 0) {
    tdm_load_wtile(W, Kdim, nblk, 0, &wtile[0][0]);
    __builtin_amdgcn_s_wait_tensorcnt(0);
  }
  __syncthreads();

  int nk = Kdim / 32;
  for (int i = 0; i < nk; i++) {
    int k = i * 32;
    if (wave == 0 && i + 1 < nk)
      tdm_load_wtile(W, Kdim, nblk, k + 32, &wtile[(i + 1) & 1][0]);   // overlap

    __builtin_prefetch(arow0 + k + 256, 0, 0);   // stream A ahead

    BFrag a0, a1;
    a0.u[0] = *(const u32x4*)(arow0 + k);
    a0.u[1] = *(const u32x4*)(arow0 + k + 16);
    a1.u[0] = *(const u32x4*)(arow1 + k);
    a1.u[1] = *(const u32x4*)(arow1 + k + 16);

    const unsigned short* wt = &wtile[i & 1][0];
    #pragma unroll
    for (int f = 0; f < 4; f++) {
      const unsigned short* wr = wt + (size_t)(wn * 64 + f * 16 + ln) * WT_ROW + 8 * hg;
      BFrag bfm;
      bfm.u[0] = ((const u32x4*)wr)[0];
      bfm.u[1] = ((const u32x4*)(wr + 16))[0];
      acc[0][f] = __builtin_amdgcn_wmma_f32_16x16x32_bf16(
          false, a0.v, false, bfm.v, (short)0, acc[0][f], false, false);
      acc[1][f] = __builtin_amdgcn_wmma_f32_16x16x32_bf16(
          false, a1.v, false, bfm.v, (short)0, acc[1][f], false, false);
    }

    __syncthreads();   // all waves done reading wtile[i&1]
    if (wave == 0 && i + 1 < nk)
      __builtin_amdgcn_s_wait_tensorcnt(0);   // next tile landed
    __syncthreads();   // release next tile to all waves
  }

  #pragma unroll
  for (int mt = 0; mt < 2; mt++) {
    #pragma unroll
    for (int f = 0; f < 4; f++) {
      #pragma unroll
      for (int r = 0; r < 8; r++) {
        int row = mbase + mt * 16 + r + 8 * hg;
        int col = nblk + wn * 64 + f * 16 + ln;
        size_t idx = (size_t)row * Ndim + col;
        float v = acc[mt][f][r];
        if (EPI == 0) {
          ((unsigned short*)out)[idx] = f2bf(v);
        } else if (EPI == 1) {
          ((float*)out)[idx] = resid[idx] + v + bias[col];
        } else {
          float t = v + bias[col];
          float ge = 0.5f * t * (1.0f + erff(t * 0.70710678118654752f));
          ((unsigned short*)out)[idx] = f2bf(ge);
        }
      }
    }
  }
}

// ---------------------------------------------------------------- Flash attention (causal), one wave per 16-query tile
__global__ __launch_bounds__(256)
void attn_fa(const unsigned short* __restrict__ q,
             const unsigned short* __restrict__ kbuf,
             const unsigned short* __restrict__ vT,
             unsigned short* __restrict__ attn) {
  __shared__ unsigned short plds[8][16 * 48];   // per-wave P staging, padded rows
  int wave = threadIdx.x >> 5, lane = threadIdx.x & 31;
  int hg = lane >> 4, ln = lane & 15;
  int task = blockIdx.x * 8 + wave;   // B*H*(T/16) = 4096 tasks
  int qt = task & 127;                // T/16 = 128
  int bh = task >> 7;                 // 0..31
  int h = bh & (Hn - 1), b = bh >> 4;
  int qbase = qt * 16;

  // Q fragments (16 x 64, two K=32 chunks)
  BFrag qa[2];
  const unsigned short* qrow =
      q + ((size_t)(b * Tn + qbase + ln)) * Cn + h * HDn;
  #pragma unroll
  for (int kc = 0; kc < 2; kc++) {
    qa[kc].u[0] = *(const u32x4*)(qrow + kc * 32 + 8 * hg);
    qa[kc].u[1] = *(const u32x4*)(qrow + kc * 32 + 16 + 8 * hg);
  }

  fvec8 o[4];
  float mrow[8], lrow[8];
  #pragma unroll
  for (int f = 0; f < 4; f++)
    #pragma unroll
    for (int r = 0; r < 8; r++) o[f][r] = 0.f;
  #pragma unroll
  for (int r = 0; r < 8; r++) { mrow[r] = -INFINITY; lrow[r] = 0.f; }

  int nchunk = qbase / 32 + 1;   // causal: only key chunks with sbase <= qmax
  for (int ch = 0; ch < nchunk; ch++) {
    int sbase = ch * 32;
    fvec8 S[2];
    #pragma unroll
    for (int st = 0; st < 2; st++) {
      #pragma unroll
      for (int r = 0; r < 8; r++) S[st][r] = 0.f;
      const unsigned short* krow =
          kbuf + ((size_t)(b * Tn + sbase + st * 16 + ln)) * Cn + h * HDn + 8 * hg;
      #pragma unroll
      for (int kc = 0; kc < 2; kc++) {
        BFrag kb;
        kb.u[0] = ((const u32x4*)(krow + kc * 32))[0];
        kb.u[1] = ((const u32x4*)(krow + kc * 32 + 16))[0];
        S[st] = __builtin_amdgcn_wmma_f32_16x16x32_bf16(
            false, qa[kc].v, false, kb.v, (short)0, S[st], false, false);
      }
    }

    // scale + causal mask + online softmax update
    float mnew[8];
    #pragma unroll
    for (int r = 0; r < 8; r++) {
      int qg = qbase + r + 8 * hg;
      #pragma unroll
      for (int st = 0; st < 2; st++) {
        int sg = sbase + st * 16 + ln;
        float v = S[st][r] * 0.125f;   // HD^-0.5
        S[st][r] = (sg > qg) ? -INFINITY : v;
      }
      float v = fmaxf(S[0][r], S[1][r]);
      #pragma unroll
      for (int off = 1; off < 16; off <<= 1) v = fmaxf(v, __shfl_xor(v, off, 32));
      mnew[r] = fmaxf(mrow[r], v);
    }
    #pragma unroll
    for (int r = 0; r < 8; r++) {
      float p0 = __expf(S[0][r] - mnew[r]);
      float p1 = __expf(S[1][r] - mnew[r]);
      float s = p0 + p1;
      #pragma unroll
      for (int off = 1; off < 16; off <<= 1) s += __shfl_xor(s, off, 32);
      float alpha = __expf(mrow[r] - mnew[r]);
      lrow[r] = lrow[r] * alpha + s;
      mrow[r] = mnew[r];
      #pragma unroll
      for (int f = 0; f < 4; f++) o[f][r] *= alpha;
      int row = r + 8 * hg;
      plds[wave][row * 48 + ln] = f2bf(p0);
      plds[wave][row * 48 + 16 + ln] = f2bf(p1);
    }
    // intra-wave LDS visibility: LDS ops are in-order per wave; wait + compiler fence
    asm volatile("s_wait_dscnt 0" ::: "memory");

    BFrag pa;   // P as A-fragment (16 x 32)
    pa.u[0] = *(const u32x4*)&plds[wave][ln * 48 + 8 * hg];
    pa.u[1] = *(const u32x4*)&plds[wave][ln * 48 + 16 + 8 * hg];

    #pragma unroll
    for (int f = 0; f < 4; f++) {
      const unsigned short* vrow =
          vT + ((size_t)(bh * HDn + f * 16 + ln)) * Tn + sbase + 8 * hg;
      BFrag vb;
      vb.u[0] = ((const u32x4*)vrow)[0];
      vb.u[1] = ((const u32x4*)(vrow + 16))[0];
      o[f] = __builtin_amdgcn_wmma_f32_16x16x32_bf16(
          false, pa.v, false, vb.v, (short)0, o[f], false, false);
    }
  }

  #pragma unroll
  for (int r = 0; r < 8; r++) {
    float rl = 1.0f / lrow[r];
    #pragma unroll
    for (int f = 0; f < 4; f++) {
      int row = b * Tn + qbase + r + 8 * hg;
      int col = h * HDn + f * 16 + ln;
      attn[(size_t)row * Cn + col] = f2bf(o[f][r] * rl);
    }
  }
}

// ---------------------------------------------------------------- launch
extern "C" void kernel_launch(void* const* d_in, const int* in_sizes, int n_in,
                              void* d_out, int out_size, void* d_ws, size_t ws_size,
                              hipStream_t stream) {
  (void)in_sizes; (void)n_in; (void)out_size; (void)ws_size;
  const float* target = (const float*)d_in[0];
  const float* wq     = (const float*)d_in[1];
  const float* wk     = (const float*)d_in[2];
  const float* wv     = (const float*)d_in[3];
  const float* wproj  = (const float*)d_in[4];
  const float* bproj  = (const float*)d_in[5];
  const float* w1     = (const float*)d_in[6];
  const float* b1     = (const float*)d_in[7];
  const float* w2     = (const float*)d_in[8];
  const float* b2     = (const float*)d_in[9];
  const float* g1     = (const float*)d_in[10];
  const float* g2     = (const float*)d_in[11];
  float* out = (float*)d_out;

  char* ws = (char*)d_ws;
  size_t off = 0;
  auto alloc = [&](size_t bytes) -> void* {
    void* p = ws + off;
    off += (bytes + 255) & ~(size_t)255;
    return p;
  };
  unsigned short* xnorm  = (unsigned short*)alloc((size_t)Mrows * Cn * 2);
  unsigned short* qb     = (unsigned short*)alloc((size_t)Mrows * Cn * 2);
  unsigned short* kb     = (unsigned short*)alloc((size_t)Mrows * Cn * 2);
  unsigned short* vb     = (unsigned short*)alloc((size_t)Mrows * Cn * 2);
  unsigned short* vTb    = (unsigned short*)alloc((size_t)Mrows * Cn * 2);
  unsigned short* attnb  = (unsigned short*)alloc((size_t)Mrows * Cn * 2);
  float*          tgt2   = (float*)alloc((size_t)Mrows * Cn * 4);
  unsigned short* yb     = (unsigned short*)alloc((size_t)Mrows * Cn * 2);
  unsigned short* hb     = (unsigned short*)alloc((size_t)Mrows * FFn * 2);
  unsigned short* wqb    = (unsigned short*)alloc((size_t)Cn * Cn * 2);
  unsigned short* wkb    = (unsigned short*)alloc((size_t)Cn * Cn * 2);
  unsigned short* wvb    = (unsigned short*)alloc((size_t)Cn * Cn * 2);
  unsigned short* wpb    = (unsigned short*)alloc((size_t)Cn * Cn * 2);
  unsigned short* w1b    = (unsigned short*)alloc((size_t)FFn * Cn * 2);
  unsigned short* w2b    = (unsigned short*)alloc((size_t)Cn * FFn * 2);

  const int CC = Cn * Cn, CF = Cn * FFn;
  cvt_f32_bf16<<<(CC + 255) / 256, 256, 0, stream>>>(wq, wqb, CC);
  cvt_f32_bf16<<<(CC + 255) / 256, 256, 0, stream>>>(wk, wkb, CC);
  cvt_f32_bf16<<<(CC + 255) / 256, 256, 0, stream>>>(wv, wvb, CC);
  cvt_f32_bf16<<<(CC + 255) / 256, 256, 0, stream>>>(wproj, wpb, CC);
  cvt_f32_bf16<<<(CF + 255) / 256, 256, 0, stream>>>(w1, w1b, CF);
  cvt_f32_bf16<<<(CF + 255) / 256, 256, 0, stream>>>(w2, w2b, CF);

  // x = rmsnorm(target, g1)
  rmsnorm_bf16<<<Mrows, 256, 0, stream>>>(target, g1, xnorm);

  // q,k,v = x @ W^T
  dim3 gQ(Mrows / 64, Cn / 256);
  gemm_bf16_tdm<0><<<gQ, 256, 0, stream>>>(xnorm, wqb, nullptr, nullptr, qb, Cn, Cn);
  gemm_bf16_tdm<0><<<gQ, 256, 0, stream>>>(xnorm, wkb, nullptr, nullptr, kb, Cn, Cn);
  gemm_bf16_tdm<0><<<gQ, 256, 0, stream>>>(xnorm, wvb, nullptr, nullptr, vb, Cn, Cn);

  transpose_v<<<(Mrows * Cn + 255) / 256, 256, 0, stream>>>(vb, vTb);

  // causal flash attention
  attn_fa<<<(Bsz * Hn * (Tn / 16)) / 8, 256, 0, stream>>>(qb, kb, vTb, attnb);

  // target2 = target + attn @ w_proj^T + b_proj
  gemm_bf16_tdm<1><<<gQ, 256, 0, stream>>>(attnb, wpb, bproj, target, tgt2, Cn, Cn);

  // y = rmsnorm(target2, g2)
  rmsnorm_bf16<<<Mrows, 256, 0, stream>>>(tgt2, g2, yb);

  // h = gelu(y @ w1^T + b1)
  dim3 gF1(Mrows / 64, FFn / 256);
  gemm_bf16_tdm<2><<<gF1, 256, 0, stream>>>(yb, w1b, b1, nullptr, hb, FFn, Cn);

  // out = target2 + h @ w2^T + b2
  gemm_bf16_tdm<1><<<gQ, 256, 0, stream>>>(hb, w2b, b2, tgt2, out, Cn, FFn);
}
